// Net_18425409700140
// MI455X (gfx1250) — compile-verified, matrix operand order checked by reading, not run on hardware
//
#include <hip/hip_runtime.h>
#include <math.h>

typedef _Float16 half_t;
typedef __attribute__((ext_vector_type(16))) _Float16 v16h;
typedef __attribute__((ext_vector_type(8)))  float    v8f;

#define NTOK   (32 * 4096)   // total tokens
#define LSEQ   4096
#define NB     32            // batch
#define CHUNK  128
#define NCHUNK (LSEQ / CHUNK)

// -------------------- CDNA5 async-DMA helpers (inline asm) ----------------
__device__ inline unsigned lds_addr32(const void* p) {
  // generic LDS pointer: low 32 bits are the wave-relative LDS byte address
  return (unsigned)(unsigned long long)p;
}
__device__ inline void async_load_b128(unsigned lds, const void* gptr) {
  asm volatile("global_load_async_to_lds_b128 %0, %1, off"
               :: "v"(lds), "v"((unsigned long long)gptr) : "memory");
}
__device__ inline void wait_async0() {
  asm volatile("s_wait_asynccnt 0x0" ::: "memory");
}

// ---------------- WMMA helpers (CDNA5 16x16x32 f16 -> f32) ----------------
// A and B share the same per-lane layout (lane&15 = M row / N col; lane>=16
// selects K+8; elems 0..7 -> K..K+7, elems 8..15 -> K+16..K+23). Staging B
// n-major (Bt[n][k] == weight row-major) makes every fragment load two
// contiguous 16-byte LDS reads.
__device__ inline v16h frag_f16(const half_t* p, int ld, int row0, int col0,
                                int lane) {
  int r = row0 + (lane & 15);
  int koff = col0 + ((lane & 16) ? 8 : 0);
  v16h f;
#pragma unroll
  for (int e = 0; e < 16; ++e) {
    int k = koff + (e & 7) + ((e & 8) << 1);
    f[e] = p[r * ld + k];
  }
  return f;
}
// Same layout but source is raw f32 in LDS (converted while packing).
__device__ inline v16h frag_f32cvt(const float* p, int ld, int col0, int lane) {
  int r = lane & 15;
  int koff = col0 + ((lane & 16) ? 8 : 0);
  v16h f;
#pragma unroll
  for (int e = 0; e < 16; ++e) {
    int k = koff + (e & 7) + ((e & 8) << 1);
    f[e] = (half_t)p[r * ld + k];
  }
  return f;
}
__device__ inline v8f wmma_f16(v16h a, v16h b, v8f c) {
  return __builtin_amdgcn_wmma_f32_16x16x32_f16(false, a, false, b, (short)0, c,
                                                false, false);
}
__device__ inline float silu_f(float v) { return v / (1.f + __expf(-v)); }

// ---------------- Stage 1: h = x @ lin_in_w.T + b  (K=64) ----------------
__global__ __launch_bounds__(32) void k_lin_in(const float* __restrict__ x,
                                               const float* __restrict__ w,
                                               const float* __restrict__ b,
                                               float* __restrict__ h) {
  __shared__ __align__(16) float  Ax[16 * 64];  // raw f32 tile via async DMA
  __shared__ __align__(16) half_t Bt[16 * 64];  // Bt[n][k] = W[n][k]
  int lane = threadIdx.x;
  long g0 = (long)blockIdx.x * 16;
  const float* src = x + g0 * 64;
#pragma unroll
  for (int i = 0; i < 8; ++i) {
    int off = (i * 32 + lane) * 4;  // float index, 16B granules
    async_load_b128(lds_addr32(&Ax[off]), src + off);
  }
  for (int idx = lane; idx < 16 * 64; idx += 32)
    Bt[idx] = (half_t)w[idx];  // weights are already n-major
  wait_async0();
  __syncthreads();
  v8f c = {};
  c = wmma_f16(frag_f32cvt(Ax, 64, 0, lane), frag_f16(Bt, 64, 0, 0, lane), c);
  c = wmma_f16(frag_f32cvt(Ax, 64, 32, lane), frag_f16(Bt, 64, 0, 32, lane), c);
  int n = lane & 15, mo = (lane & 16) ? 8 : 0;
  float bias = b[n];
#pragma unroll
  for (int r = 0; r < 8; ++r)
    h[(g0 + r + mo) * 16 + n] = c[r] + bias;
}

// ------- Stage 2a: rmsnorm + in_proj (K=16 padded to 32, 64 out cols) -----
__global__ __launch_bounds__(32) void k_inproj(const float* __restrict__ h,
                                               const float* __restrict__ normw,
                                               const float* __restrict__ w,
                                               float* __restrict__ xin,
                                               float* __restrict__ zout) {
  __shared__ __align__(16) half_t Ah[16 * 32];
  __shared__ __align__(16) half_t Bt[64 * 32];  // Bt[n][k], K zero-padded
  int lane = threadIdx.x;
  long g0 = (long)blockIdx.x * 16;
  if (lane < 16) {
    float v[16], ss = 0.f;
#pragma unroll
    for (int k = 0; k < 16; ++k) { v[k] = h[(g0 + lane) * 16 + k]; ss += v[k] * v[k]; }
    float inv = rsqrtf(ss * (1.0f / 16.0f) + 1e-5f);
#pragma unroll
    for (int k = 0; k < 16; ++k) {
      Ah[lane * 32 + k] = (half_t)(v[k] * inv * normw[k]);
      Ah[lane * 32 + 16 + k] = (half_t)0.f;
    }
  }
  for (int idx = lane; idx < 64 * 32; idx += 32) {
    int n = idx >> 5, k = idx & 31;
    Bt[idx] = (k < 16) ? (half_t)w[n * 16 + k] : (half_t)0.f;
  }
  __syncthreads();
  v16h a = frag_f16(Ah, 32, 0, 0, lane);
  int n = lane & 15, mo = (lane & 16) ? 8 : 0;
#pragma unroll
  for (int nf = 0; nf < 4; ++nf) {
    v8f c = {};
    c = wmma_f16(a, frag_f16(Bt, 32, nf * 16, 0, lane), c);
    int col = nf * 16 + n;
    float* dst = (col < 32) ? xin : zout;
    int cc = col & 31;
#pragma unroll
    for (int r = 0; r < 8; ++r)
      dst[(g0 + r + mo) * 32 + cc] = c[r];
  }
}

// -- Stage 2b: causal dwconv(K=4)+SiLU, x_proj (K=32 exact), delta, B/C ----
__global__ __launch_bounds__(32) void k_conv_xproj(
    const float* __restrict__ xin, const float* __restrict__ convw,
    const float* __restrict__ convb, const float* __restrict__ xpw,
    const float* __restrict__ dtw, const float* __restrict__ dtb,
    float* __restrict__ xc, float* __restrict__ dlt,
    float* __restrict__ Bws, float* __restrict__ Cws) {
  __shared__ __align__(16) half_t Ah[16 * 32];
  __shared__ __align__(16) half_t Bt[48 * 32];  // x_proj_w rows, padded to 48
  __shared__ __align__(16) float dbc[16 * 48];
  int lane = threadIdx.x;  // = channel
  long g0 = (long)blockIdx.x * 16;
  int t0 = (int)(g0 % LSEQ);
  long bb = g0 - t0;  // batch base token
  float w0 = convw[lane * 4 + 0], w1 = convw[lane * 4 + 1];
  float w2 = convw[lane * 4 + 2], w3 = convw[lane * 4 + 3];
  float cb = convb[lane];
  // sliding conv window
  float xm3 = (t0 >= 3) ? xin[(bb + t0 - 3) * 32 + lane] : 0.f;
  float xm2 = (t0 >= 2) ? xin[(bb + t0 - 2) * 32 + lane] : 0.f;
  float xm1 = (t0 >= 1) ? xin[(bb + t0 - 1) * 32 + lane] : 0.f;
#pragma unroll
  for (int i = 0; i < 16; ++i) {
    float x0 = xin[(bb + t0 + i) * 32 + lane];
    float acc = cb + w0 * xm3 + w1 * xm2 + w2 * xm1 + w3 * x0;
    float s = silu_f(acc);
    xc[(g0 + i) * 32 + lane] = s;
    Ah[i * 32 + lane] = (half_t)s;
    xm3 = xm2; xm2 = xm1; xm1 = x0;
  }
  for (int idx = lane; idx < 48 * 32; idx += 32)
    Bt[idx] = (idx < 33 * 32) ? (half_t)xpw[idx] : (half_t)0.f;  // n-major copy
  __syncthreads();
  v16h a = frag_f16(Ah, 32, 0, 0, lane);
  int nn = lane & 15, mo = (lane & 16) ? 8 : 0;
#pragma unroll
  for (int nf = 0; nf < 3; ++nf) {
    v8f c = {};
    c = wmma_f16(a, frag_f16(Bt, 32, nf * 16, 0, lane), c);
#pragma unroll
    for (int r = 0; r < 8; ++r)
      dbc[(r + mo) * 48 + nf * 16 + nn] = c[r];
  }
  __syncthreads();
  float dw = dtw[lane], db = dtb[lane];
#pragma unroll
  for (int i = 0; i < 16; ++i) {
    float u = dbc[i * 48] * dw + db;  // DT_RANK == 1
    float sp = (u > 20.f) ? u : __logf(1.f + __expf(u));
    dlt[(g0 + i) * 32 + lane] = sp;
  }
  for (int idx = lane; idx < 16 * 16; idx += 32) {
    int i = idx >> 4, n = idx & 15;
    Bws[(g0 + i) * 16 + n] = dbc[i * 48 + 1 + n];
    Cws[(g0 + i) * 16 + n] = dbc[i * 48 + 17 + n];
  }
}

// ------- Stage 3: chunked parallel scan. Pass1: local scan + aggregates ---
__global__ __launch_bounds__(512) void k_scan1(
    const float* __restrict__ dlt, const float* __restrict__ Bws,
    const float* __restrict__ xc, const float* __restrict__ Alog,
    float* __restrict__ qbuf, float* __restrict__ Pbuf) {
  int tid = threadIdx.x;
  int ch = tid >> 4, n = tid & 15;
  int chunk = blockIdx.x, batch = blockIdx.y;
  long base = (long)batch * LSEQ + (long)chunk * CHUNK;
  float A = -__expf(Alog[ch * 16 + n]);
  float h = 0.f, sd = 0.f;
  for (int t = 0; t < CHUNK; ++t) {
    long g = base + t;
    __builtin_prefetch(&dlt[(g + 16) * 32 + ch], 0, 0);
    float dl = dlt[g * 32 + ch];
    float bv = Bws[g * 16 + n];
    float xv = xc[g * 32 + ch];
    h = __expf(dl * A) * h + dl * bv * xv;
    sd += dl;
  }
  long idx = ((long)(batch * NCHUNK + chunk)) * 512 + tid;
  qbuf[idx] = h;
  Pbuf[idx] = __expf(sd * A);  // prod_t exp(dl_t*A) = exp(A * sum dl)
}

// Pass2: serial combine over 32 chunk aggregates per batch
__global__ __launch_bounds__(512) void k_combine(const float* __restrict__ Pbuf,
                                                 const float* __restrict__ qbuf,
                                                 float* __restrict__ h0buf) {
  int tid = threadIdx.x;
  int batch = blockIdx.x;
  float h0 = 0.f;
  for (int c = 0; c < NCHUNK; ++c) {
    long idx = ((long)(batch * NCHUNK + c)) * 512 + tid;
    h0buf[idx] = h0;
    h0 = Pbuf[idx] * h0 + qbuf[idx];
  }
}

// Pass3: re-scan with correct h0, emit y = C.h + D*x (shfl reduce over states)
__global__ __launch_bounds__(512) void k_scan2(
    const float* __restrict__ dlt, const float* __restrict__ Bws,
    const float* __restrict__ Cws, const float* __restrict__ xc,
    const float* __restrict__ Alog, const float* __restrict__ Dp,
    const float* __restrict__ h0buf, float* __restrict__ y) {
  int tid = threadIdx.x;
  int ch = tid >> 4, n = tid & 15;
  int chunk = blockIdx.x, batch = blockIdx.y;
  long base = (long)batch * LSEQ + (long)chunk * CHUNK;
  float A = -__expf(Alog[ch * 16 + n]);
  long cidx = ((long)(batch * NCHUNK + chunk)) * 512 + tid;
  float h = h0buf[cidx];
  float Dv = Dp[ch];
  for (int t = 0; t < CHUNK; ++t) {
    long g = base + t;
    __builtin_prefetch(&dlt[(g + 16) * 32 + ch], 0, 0);
    float dl = dlt[g * 32 + ch];
    float bv = Bws[g * 16 + n];
    float xv = xc[g * 32 + ch];
    float cv = Cws[g * 16 + n];
    h = __expf(dl * A) * h + dl * bv * xv;
    float p = h * cv;
    p += __shfl_xor(p, 1, 16);
    p += __shfl_xor(p, 2, 16);
    p += __shfl_xor(p, 4, 16);
    p += __shfl_xor(p, 8, 16);
    if (n == 0) y[g * 32 + ch] = p + Dv * xv;
  }
}

// ------- Stage 4: out = (y * silu(z)) @ out_proj.T + h (K=32 exact) -------
__global__ __launch_bounds__(32) void k_outproj(const float* __restrict__ y,
                                                const float* __restrict__ z,
                                                const float* __restrict__ w,
                                                float* __restrict__ h) {
  __shared__ __align__(16) half_t Ah[16 * 32];
  __shared__ __align__(16) half_t Bt[16 * 32];  // out_proj_w is n-major already
  int lane = threadIdx.x;
  long g0 = (long)blockIdx.x * 16;
  for (int idx = lane; idx < 16 * 32; idx += 32) {
    float zv = z[g0 * 32 + idx];
    Ah[idx] = (half_t)(y[g0 * 32 + idx] * silu_f(zv));
  }
  for (int idx = lane; idx < 16 * 32; idx += 32)
    Bt[idx] = (half_t)w[idx];
  __syncthreads();
  v8f c = {};
  c = wmma_f16(frag_f16(Ah, 32, 0, 0, lane), frag_f16(Bt, 32, 0, 0, lane), c);
  int n = lane & 15, mo = (lane & 16) ? 8 : 0;
#pragma unroll
  for (int r = 0; r < 8; ++r) {
    long gi = (g0 + r + mo) * 16 + n;
    h[gi] = h[gi] + c[r];  // residual
  }
}

// ------------------- Stage 5: out = h @ lin_out.T + b ---------------------
__global__ void k_final(const float* __restrict__ h, const float* __restrict__ w,
                        const float* __restrict__ b, float* __restrict__ out) {
  long g = (long)blockIdx.x * blockDim.x + threadIdx.x;
  if (g >= NTOK) return;
  const float4* hv = (const float4*)(h + g * 16);
  const float4* wv = (const float4*)w;
  float s = b[0];
#pragma unroll
  for (int q = 0; q < 4; ++q) {
    float4 hq = hv[q], wq = wv[q];
    s += hq.x * wq.x + hq.y * wq.y + hq.z * wq.z + hq.w * wq.w;
  }
  out[g] = s;
}

extern "C" void kernel_launch(void* const* d_in, const int* in_sizes, int n_in,
                              void* d_out, int out_size, void* d_ws, size_t ws_size,
                              hipStream_t stream) {
  (void)in_sizes; (void)n_in; (void)out_size; (void)ws_size;
  const float* x         = (const float*)d_in[0];
  const float* lin_in_w  = (const float*)d_in[1];
  const float* lin_in_b  = (const float*)d_in[2];
  const float* lin_out_w = (const float*)d_in[23];
  const float* lin_out_b = (const float*)d_in[24];

  float* ws  = (float*)d_ws;
  float* h   = ws;
  float* xin = h + (size_t)NTOK * 16;
  float* z   = xin + (size_t)NTOK * 32;
  float* xc  = z + (size_t)NTOK * 32;
  float* dlt = xc + (size_t)NTOK * 32;
  float* Bws = dlt + (size_t)NTOK * 32;
  float* Cws = Bws + (size_t)NTOK * 16;
  float* yws = Cws + (size_t)NTOK * 16;
  float* qb  = yws + (size_t)NTOK * 32;
  float* Pb  = qb + (size_t)NB * NCHUNK * 512;
  float* h0b = Pb + (size_t)NB * NCHUNK * 512;

  const int tiles = NTOK / 16;  // 8192 one-wave tiles
  k_lin_in<<<tiles, 32, 0, stream>>>(x, lin_in_w, lin_in_b, h);

  for (int layer = 0; layer < 2; ++layer) {
    int o = 3 + layer * 10;
    const float* normw = (const float*)d_in[o + 0];
    const float* ipw   = (const float*)d_in[o + 1];
    const float* cw    = (const float*)d_in[o + 2];
    const float* cb    = (const float*)d_in[o + 3];
    const float* xpw   = (const float*)d_in[o + 4];
    const float* dtw   = (const float*)d_in[o + 5];
    const float* dtb   = (const float*)d_in[o + 6];
    const float* Alog  = (const float*)d_in[o + 7];
    const float* Dp    = (const float*)d_in[o + 8];
    const float* opw   = (const float*)d_in[o + 9];

    k_inproj<<<tiles, 32, 0, stream>>>(h, normw, ipw, xin, z);
    k_conv_xproj<<<tiles, 32, 0, stream>>>(xin, cw, cb, xpw, dtw, dtb,
                                           xc, dlt, Bws, Cws);
    dim3 sg(NCHUNK, NB);
    k_scan1<<<sg, 512, 0, stream>>>(dlt, Bws, xc, Alog, qb, Pb);
    k_combine<<<NB, 512, 0, stream>>>(Pb, qb, h0b);
    k_scan2<<<sg, 512, 0, stream>>>(dlt, Bws, Cws, xc, Alog, Dp, h0b, yws);
    k_outproj<<<tiles, 32, 0, stream>>>(yws, z, opw, h);
  }

  k_final<<<(NTOK + 255) / 256, 256, 0, stream>>>(h, lin_out_w, lin_out_b,
                                                  (float*)d_out);
}
